// RelativePositionTransformerS4_11605001634067
// MI455X (gfx1250) — matrix operand order repaired and checked by reading, not compiled, not run on hardware
//
#include <hip/hip_runtime.h>
#include <hip/hip_bf16.h>
#include <math.h>
#include <stdint.h>

typedef __bf16 bf16_t;
typedef __attribute__((ext_vector_type(16))) __bf16 v16bf;
typedef __attribute__((ext_vector_type(8)))  __bf16 v8bf;
typedef __attribute__((ext_vector_type(8)))  float   v8f;

#define LYR 6
#define Bn  16
#define Hc  384
#define H2  768
#define Tn  2048
#define NST 32
#define HF  1536

// 1 = use GLOBAL_LOAD_ASYNC_TO_LDS_B128 double-buffered pipeline (ASYNCcnt)
// 0 = synchronous global->VGPR->LDS fallback
#define USE_ASYNC_COPY 1

#if USE_ASYNC_COPY
__device__ __forceinline__ void async_ld16(uint32_t lds_addr, const void* gaddr) {
  // DMA 16 bytes global -> LDS, tracked with ASYNCcnt (no VGPR staging)
  asm volatile("global_load_async_to_lds_b128 %0, %1, off"
               :: "v"(lds_addr), "v"(gaddr) : "memory");
}
__device__ __forceinline__ void async_wait0() {
  asm volatile("s_wait_asynccnt 0x0" ::: "memory");
}
#endif

// ---------------------------------------------------------------- utilities
__global__ __launch_bounds__(256) void cvt_f32_bf16(const float* __restrict__ s,
                                                    bf16_t* __restrict__ d, size_t n) {
  size_t i = (size_t)blockIdx.x * blockDim.x + threadIdx.x;
  if (i < n) d[i] = (bf16_t)s[i];
}

// ------------------------------------------------- S4D discretization (tiny)
__global__ __launch_bounds__(256) void s4_precompute(
    const float* __restrict__ logdt, const float* __restrict__ Are,
    const float* __restrict__ Aim,   const float* __restrict__ Cre,
    const float* __restrict__ Cim,
    float* __restrict__ wre, float* __restrict__ wim,
    float* __restrict__ cdre, float* __restrict__ cdim,
    float* __restrict__ wcre, float* __restrict__ wcim) {
  int i = blockIdx.x * blockDim.x + threadIdx.x;
  if (i >= Hc * NST) return;
  int h = i / NST;
  float dt = __expf(logdt[h]);
  float ar = Are[i], ai = Aim[i];
  float dr = ar * dt, di = ai * dt;
  float e  = __expf(dr);
  float wr = e * __cosf(di), wi = e * __sinf(di);       // w = exp(dt*A)
  float emr = wr - 1.f, emi = wi;                       // Cd = C*(exp(dtA)-1)/A
  float nr = Cre[i] * emr - Cim[i] * emi;
  float ni = Cre[i] * emi + Cim[i] * emr;
  float inv = 1.f / (ar * ar + ai * ai);
  cdre[i] = (nr * ar + ni * ai) * inv;
  cdim[i] = (ni * ar - nr * ai) * inv;
  wre[i] = wr; wim[i] = wi;
  float sr = wr, si = wi;                               // w^256 by 8 squarings
#pragma unroll
  for (int k = 0; k < 8; k++) { float t = sr * sr - si * si; si = 2.f * sr * si; sr = t; }
  wcre[i] = sr; wcim[i] = si;
}

// ---------------------------- chunked diagonal-SSM scan + skip + exact GELU
__global__ __launch_bounds__(256) void s4_scan(
    const float* __restrict__ xin, const float* __restrict__ xmask,
    const float* __restrict__ wre, const float* __restrict__ wim,
    const float* __restrict__ cdre, const float* __restrict__ cdim,
    const float* __restrict__ wcre, const float* __restrict__ wcim,
    const float* __restrict__ Dp,
    float* __restrict__ xout, bf16_t* __restrict__ gout) {
  __shared__ float xs[Tn];
  __shared__ float ser[8][NST], sei[8][NST];
  int bh = blockIdx.x, b = bh / Hc, h = bh % Hc;
  int tid = threadIdx.x, wid = tid >> 5, lane = tid & 31;
  size_t row = (size_t)bh * Tn;
  const float* xr = xin + row;
  const float* mk = xmask + (size_t)b * Tn;
  for (int i = tid; i < Tn; i += 256) {
    float v = xr[i] * mk[i];
    xs[i] = v;
    xout[row + i] = v;                       // masked x for the residual path
  }
  __syncthreads();
  int pi = h * NST + lane;
  float wr = wre[pi], wi = wim[pi], cr = cdre[pi], ci = cdim[pi], Dh = Dp[h];
  int base = wid * 256;
  float sr = 0.f, si = 0.f;                  // pass 1: local end-state
  for (int l = 0; l < 256; l++) {
    float xl = xs[base + l];
    float t  = fmaf(wr, sr, fmaf(-wi, si, xl));
    si = fmaf(wr, si, wi * sr);
    sr = t;
  }
  ser[wid][lane] = sr; sei[wid][lane] = si;
  __syncthreads();
  float wcr = wcre[pi], wci = wcim[pi], Sr = 0.f, Si = 0.f;
  for (int j = 0; j < wid; j++) {            // incoming state via w^256 powers
    float t = fmaf(wcr, Sr, -wci * Si) + ser[j][lane];
    Si = fmaf(wcr, Si, wci * Sr) + sei[j][lane];
    Sr = t;
  }
  sr = Sr; si = Si;                          // pass 2: y = 2*Re(Cd.s) + D*x
  float yreg[8];
  for (int l = 0; l < 256; l++) {
    float xl = xs[base + l];
    float t  = fmaf(wr, sr, fmaf(-wi, si, xl));
    si = fmaf(wr, si, wi * sr);
    sr = t;
    float p = cr * sr - ci * si;
#pragma unroll
    for (int off = 16; off > 0; off >>= 1) p += __shfl_xor(p, off, 32);
    float y = 2.f * p + Dh * xl;
    float g = 0.5f * y * (1.f + erff(y * 0.70710678f));
    if ((l & 31) == lane) yreg[l >> 5] = g;
  }
  size_t tb = (size_t)b * Tn + base;         // transposed store (b,t,h)
#pragma unroll
  for (int i = 0; i < 8; i++)
    gout[(tb + i * 32 + lane) * Hc + h] = (bf16_t)yreg[i];
}

// --------------------------------------------- bf16 WMMA GEMM (batched in z)
// C[b](MxN) = A(MxK) * Xt[b](NxK)^T + bias
// LDS tiles in WMMA fragment layout; double-buffered async global->LDS DMA.
// epi: 0 = f32 out (b,M,N) ; 1 = relu -> bf16 out TRANSPOSED (b,N,M)
__global__ __launch_bounds__(256) void gemm_wmma_bf16(
    const bf16_t* __restrict__ A, const bf16_t* __restrict__ Xt,
    const float* __restrict__ bias,
    float* __restrict__ Cf, bf16_t* __restrict__ Cb,
    int M, int N, int K, int epi) {
  __shared__ v16bf Asf[2][2][4][32];         // [buf][k-slice][wave-row][lane] 16 KB
  __shared__ v16bf Bsf[2][2][8][32];         // [buf][k-slice][n-subtile][lane] 32 KB
  int b = blockIdx.z;
  int bm0 = blockIdx.y * 64, bn0 = blockIdx.x * 128;
  const bf16_t* Xp = Xt + (size_t)b * N * K;
  int tid = threadIdx.x, wid = tid >> 5, lane = tid & 31;
  int wrow = wid & 3, wcol = wid >> 2;       // 4x2 waves -> 64x128 tile
  v8f acc[4];
#pragma unroll
  for (int j = 0; j < 4; j++)
#pragma unroll
    for (int i = 0; i < 8; i++) acc[j][i] = 0.f;

  auto fill = [&](int k0, int buf) {
    // A fragments: slot j<8 -> k=half*8+j ; j>=8 -> k=half*8+16+(j-8)
#pragma unroll
    for (int q = 0; q < 2; q++) {            // 512 16B chunks, 2/thread
      int idx = tid + q * 256;
      int ks = idx >> 8, rem = idx & 255;
      int wr = rem >> 6, l = (rem >> 1) & 31, c = rem & 1;
      int am = bm0 + wr * 16 + (l & 15);
      int ak = k0 + ks * 32 + ((l >> 4) * 8) + c * 16;
#if USE_ASYNC_COPY
      async_ld16((uint32_t)(size_t)(((v8bf*)&Asf[buf][ks][wr][l]) + c),
                 &A[(size_t)am * K + ak]);
#else
      ((v8bf*)&Asf[buf][ks][wr][l])[c] = *(const v8bf*)&A[(size_t)am * K + ak];
#endif
    }
    // B fragments: slot j<8 -> k=half*8+j ; j>=8 -> k=half*8+16+(j-8)
#pragma unroll
    for (int q = 0; q < 4; q++) {            // 1024 16B chunks, 4/thread
      int idx = tid + q * 256;
      int ks = idx >> 9, rem = idx & 511;
      int s = rem >> 6, l = (rem >> 1) & 31, c = rem & 1;
      int n = bn0 + s * 16 + (l & 15);
      int k = k0 + ks * 32 + ((l >> 4) * 8) + c * 16;
#if USE_ASYNC_COPY
      async_ld16((uint32_t)(size_t)(((v8bf*)&Bsf[buf][ks][s][l]) + c),
                 &Xp[(size_t)n * K + k]);
#else
      ((v8bf*)&Bsf[buf][ks][s][l])[c] = *(const v8bf*)&Xp[(size_t)n * K + k];
#endif
    }
  };

  int nk = K >> 6;                           // K-step = 64
#if USE_ASYNC_COPY
  fill(0, 0);
  async_wait0();
  __syncthreads();
  for (int kt = 0; kt < nk; kt++) {
    int buf = kt & 1;
    if (kt + 1 < nk) fill((kt + 1) << 6, buf ^ 1);   // DMA next tile during compute
#pragma unroll
    for (int ks = 0; ks < 2; ks++) {
      v16bf af = Asf[buf][ks][wrow][lane];
#pragma unroll
      for (int jt = 0; jt < 4; jt++) {
        v16bf bfv = Bsf[buf][ks][wcol * 4 + jt][lane];
        acc[jt] = __builtin_amdgcn_wmma_f32_16x16x32_bf16(
            false, af, false, bfv, (short)0, acc[jt], false, false);
      }
    }
    async_wait0();
    __syncthreads();
  }
#else
  for (int kt = 0; kt < nk; kt++) {
    fill(kt << 6, 0);
    __syncthreads();
#pragma unroll
    for (int ks = 0; ks < 2; ks++) {
      v16bf af = Asf[0][ks][wrow][lane];
#pragma unroll
      for (int jt = 0; jt < 4; jt++) {
        v16bf bfv = Bsf[0][ks][wcol * 4 + jt][lane];
        acc[jt] = __builtin_amdgcn_wmma_f32_16x16x32_bf16(
            false, af, false, bfv, (short)0, acc[jt], false, false);
      }
    }
    __syncthreads();
  }
#endif
  int half = lane >> 4, r = lane & 15;
#pragma unroll
  for (int jt = 0; jt < 4; jt++) {
    int nc = bn0 + wcol * 64 + jt * 16 + r;
    int m0 = bm0 + wrow * 16 + half * 8;     // C/D layout: VGPR i -> M=half*8+i
    if (epi == 0) {
#pragma unroll
      for (int i = 0; i < 8; i++)
        Cf[(size_t)b * M * N + (size_t)(m0 + i) * N + nc] = acc[jt][i] + bias[m0 + i];
    } else {                                 // relu -> bf16, transposed (b,N,M)
      v8bf pk;
#pragma unroll
      for (int i = 0; i < 8; i++)
        pk[i] = (bf16_t)fmaxf(acc[jt][i] + bias[m0 + i], 0.f);
      *(v8bf*)&Cb[((size_t)b * N + nc) * M + m0] = pk;
    }
  }
}

// --------------------------- GLU + residual + channel-LayerNorm (eps 1e-4)
__global__ __launch_bounds__(256) void glu_res_ln(
    float* __restrict__ z, const float* __restrict__ xcur,
    const float* __restrict__ xmask,
    const float* __restrict__ gam, const float* __restrict__ bet,
    float* __restrict__ xout, bf16_t* __restrict__ bxout) {
  int t = blockIdx.x * 256 + threadIdx.x, b = blockIdx.y;
  float mv = xmask[(size_t)b * Tn + t];
  size_t zb = (size_t)b * H2 * Tn, xb = (size_t)b * Hc * Tn;
  float sum = 0.f, ss = 0.f;
  for (int h = 0; h < Hc; h++) {
    float a = z[zb + (size_t)h * Tn + t];
    float g = z[zb + (size_t)(h + Hc) * Tn + t];
    float y = a * (1.f / (1.f + __expf(-g)));
    float xn = xcur[xb + (size_t)h * Tn + t] + y;
    z[zb + (size_t)h * Tn + t] = xn;         // stash for pass 2
    sum += xn; ss += xn * xn;
  }
  float m = sum * (1.f / Hc);
  float rs = rsqrtf(ss * (1.f / Hc) - m * m + 1e-4f);
  bf16_t* bxrow = bxout + ((size_t)b * Tn + t) * Hc;
  v8bf pk;
  for (int h = 0; h < Hc; h++) {
    float xn = z[zb + (size_t)h * Tn + t];
    float o = (xn - m) * rs * gam[h] + bet[h];
    xout[xb + (size_t)h * Tn + t] = o;
    pk[h & 7] = (bf16_t)(o * mv);
    if ((h & 7) == 7) *(v8bf*)&bxrow[h - 7] = pk;
  }
}

__global__ __launch_bounds__(256) void res_ln2(
    float* __restrict__ o, const float* __restrict__ xcur,
    const float* __restrict__ xmask,
    const float* __restrict__ gam, const float* __restrict__ bet,
    float* __restrict__ xout, int maskOut) {
  int t = blockIdx.x * 256 + threadIdx.x, b = blockIdx.y;
  float mv = xmask[(size_t)b * Tn + t];
  size_t xb = (size_t)b * Hc * Tn;
  float sum = 0.f, ss = 0.f;
  for (int h = 0; h < Hc; h++) {
    size_t idx = xb + (size_t)h * Tn + t;
    float xn = xcur[idx] + o[idx] * mv;
    o[idx] = xn;
    sum += xn; ss += xn * xn;
  }
  float m = sum * (1.f / Hc);
  float rs = rsqrtf(ss * (1.f / Hc) - m * m + 1e-4f);
  for (int h = 0; h < Hc; h++) {
    size_t idx = xb + (size_t)h * Tn + t;
    float v = (o[idx] - m) * rs * gam[h] + bet[h];
    if (maskOut) v *= mv;
    xout[idx] = v;
  }
}

// ------------------------------------------------------------------- driver
extern "C" void kernel_launch(void* const* d_in, const int* in_sizes, int n_in,
                              void* d_out, int out_size, void* d_ws, size_t ws_size,
                              hipStream_t stream) {
  (void)in_sizes; (void)n_in; (void)out_size; (void)ws_size;
  const float* x     = (const float*)d_in[0];
  const float* xmask = (const float*)d_in[1];
  const float* logdt = (const float*)d_in[2];
  const float* Are   = (const float*)d_in[3];
  const float* Aim   = (const float*)d_in[4];
  const float* Cre   = (const float*)d_in[5];
  const float* Cim   = (const float*)d_in[6];
  const float* Dp    = (const float*)d_in[7];
  const float* Wout  = (const float*)d_in[8];
  const float* bout  = (const float*)d_in[9];
  const float* g1    = (const float*)d_in[10];
  const float* be1   = (const float*)d_in[11];
  const float* W1    = (const float*)d_in[12];
  const float* bf1   = (const float*)d_in[13];
  const float* W2    = (const float*)d_in[14];
  const float* bf2   = (const float*)d_in[15];
  const float* g2    = (const float*)d_in[16];
  const float* be2   = (const float*)d_in[17];
  float* out = (float*)d_out;                    // doubles as f32 activation buffer

  char* ws = (char*)d_ws;
  size_t off = 0;
  auto alloc = [&](size_t bytes) -> void* {
    void* p = ws + off; off += (bytes + 255) & ~(size_t)255; return p;
  };
  bf16_t* bx   = (bf16_t*)alloc((size_t)Bn * Tn * Hc * sizeof(bf16_t));  // (b,t,h)
  float*  z32  = (float*) alloc((size_t)Bn * H2 * Tn * sizeof(float));
  bf16_t* hb   = (bf16_t*)alloc((size_t)Bn * Tn * HF * sizeof(bf16_t));  // (b,t,hf)
  bf16_t* wO   = (bf16_t*)alloc((size_t)LYR * H2 * Hc * sizeof(bf16_t));
  bf16_t* wF1  = (bf16_t*)alloc((size_t)LYR * HF * Hc * sizeof(bf16_t));
  bf16_t* wF2  = (bf16_t*)alloc((size_t)LYR * Hc * HF * sizeof(bf16_t));
  float* wre  = (float*)alloc(Hc * NST * 4);
  float* wim  = (float*)alloc(Hc * NST * 4);
  float* cdre = (float*)alloc(Hc * NST * 4);
  float* cdim = (float*)alloc(Hc * NST * 4);
  float* wcre = (float*)alloc(Hc * NST * 4);
  float* wcim = (float*)alloc(Hc * NST * 4);

  {                                              // weights f32 -> bf16 once
    size_t n;
    n = (size_t)LYR * H2 * Hc;
    cvt_f32_bf16<<<(unsigned)((n + 255) / 256), 256, 0, stream>>>(Wout, wO, n);
    n = (size_t)LYR * HF * Hc;
    cvt_f32_bf16<<<(unsigned)((n + 255) / 256), 256, 0, stream>>>(W1, wF1, n);
    n = (size_t)LYR * Hc * HF;
    cvt_f32_bf16<<<(unsigned)((n + 255) / 256), 256, 0, stream>>>(W2, wF2, n);
  }

  for (int i = 0; i < LYR; i++) {
    s4_precompute<<<(Hc * NST + 255) / 256, 256, 0, stream>>>(
        logdt + (size_t)i * Hc, Are + (size_t)i * Hc * NST, Aim + (size_t)i * Hc * NST,
        Cre + (size_t)i * Hc * NST, Cim + (size_t)i * Hc * NST,
        wre, wim, cdre, cdim, wcre, wcim);

    const float* xin = (i == 0) ? x : out;
    s4_scan<<<Bn * Hc, 256, 0, stream>>>(xin, xmask, wre, wim, cdre, cdim,
                                         wcre, wcim, Dp + (size_t)i * Hc, out, bx);

    // z = Wout @ gelu(y)  (768 x 2048 x 384, batched over B)
    gemm_wmma_bf16<<<dim3(Tn / 128, H2 / 64, Bn), 256, 0, stream>>>(
        wO + (size_t)i * H2 * Hc, bx, bout + (size_t)i * H2, z32, nullptr,
        H2, Tn, Hc, 0);

    glu_res_ln<<<dim3(Tn / 256, Bn), 256, 0, stream>>>(
        z32, out, xmask, g1 + (size_t)i * Hc, be1 + (size_t)i * Hc, out, bx);

    // FFN1: relu(W1 @ x) -> bf16 transposed (1536 x 2048 x 384)
    gemm_wmma_bf16<<<dim3(Tn / 128, HF / 64, Bn), 256, 0, stream>>>(
        wF1 + (size_t)i * HF * Hc, bx, bf1 + (size_t)i * HF, nullptr, hb,
        HF, Tn, Hc, 1);

    // FFN2: W2 @ h -> f32 (384 x 2048 x 1536)
    gemm_wmma_bf16<<<dim3(Tn / 128, Hc / 64, Bn), 256, 0, stream>>>(
        wF2 + (size_t)i * Hc * HF, hb, bf2 + (size_t)i * Hc, z32, nullptr,
        Hc, Tn, HF, 0);

    res_ln2<<<dim3(Tn / 256, Bn), 256, 0, stream>>>(
        z32, out, xmask, g2 + (size_t)i * Hc, be2 + (size_t)i * Hc, out,
        (i == LYR - 1) ? 1 : 0);
  }
}
